// HCLSTM_10737418240541
// MI455X (gfx1250) — matrix operand order repaired and checked by reading, not compiled, not run on hardware
//
#include <hip/hip_runtime.h>
#include <hip/hip_bf16.h>
#include <math.h>

#define BATCH 16
#define TLEN  256
#define HID   512
#define VOCAB 32000
#define MROWS (BATCH * TLEN)   // 4096
#define G4    (4 * HID)        // 2048
#define HPAD  520              // padded LDS row stride for lstm h (bf16 elems)
#define APAD  40               // padded LDS row stride for gemm A stage (bf16)

typedef __attribute__((ext_vector_type(16))) __bf16 v16bf;
typedef __attribute__((ext_vector_type(8)))  __bf16 v8bf;
typedef __attribute__((ext_vector_type(8)))  float  v8f;
typedef __attribute__((address_space(3)))    __bf16 lds_bf16;

// ---------------------------------------------------------------------------
// Fragment helpers (wave32 WMMA, K order used by this file, 16x16x32 bf16)
// A tile (16M x 32K): lane<16 -> M=lane, halves0-15=K0..15
//                     lane>=16 -> M=lane-16, K16..31
// B tile (32K x 16N): lane holds column N=lane&15; K base 0 (lane<16) / 16,
//                     16 contiguous K values -> one contiguous 32B row of W
//                     when computing out = A * W^T (B[k][n] = W[n][k]).
// ---------------------------------------------------------------------------
__device__ __forceinline__ v16bf make_frag(v8bf lo, v8bf hi) {
  return __builtin_shufflevector(lo, hi, 0,1,2,3,4,5,6,7,8,9,10,11,12,13,14,15);
}

__device__ __forceinline__ v16bf load_B_frag(const __bf16* __restrict__ base,
                                             int ld, int n0, int k0, int lane) {
  int n  = n0 + (lane & 15);
  int kb = (lane < 16) ? 0 : 16;
  const __bf16* p = base + (size_t)n * ld + k0 + kb;
  return *reinterpret_cast<const v16bf*>(p);
}

#define WMMA_BF16(a, b, c) \
  __builtin_amdgcn_wmma_f32_16x16x32_bf16(false, (a), false, (b), (short)0, (c), false, false)

// Async global->LDS 16B copy (CDNA5, ASYNCcnt-tracked)
__device__ __forceinline__ void async_copy_b128(unsigned lds_off, const __bf16* g) {
  asm volatile("global_load_async_to_lds_b128 %0, %1, off"
               :: "v"(lds_off), "v"((unsigned long long)(uintptr_t)g)
               : "memory");
}

// ---------------------------------------------------------------------------
// Weight prep: bf16 conversions + gate concat.  Wh[n=g*512+j][k] = Wg[j][k],
// Wx[n][k] = Wg[j][512+k], biascat[n] = bg[j], fcW16[v][k] = fcW[v][k].
// ---------------------------------------------------------------------------
__global__ __launch_bounds__(256) void prep_kernel(
    const float* __restrict__ Wf, const float* __restrict__ Wi,
    const float* __restrict__ Wc, const float* __restrict__ Wo,
    const float* __restrict__ bf_, const float* __restrict__ bi_,
    const float* __restrict__ bc_, const float* __restrict__ bo_,
    const float* __restrict__ fcW,
    __bf16* __restrict__ Wh16, __bf16* __restrict__ Wx16,
    float* __restrict__ biascat, __bf16* __restrict__ fcW16) {
  size_t i = (size_t)blockIdx.x * blockDim.x + threadIdx.x;
  if (i < (size_t)VOCAB * HID) fcW16[i] = (__bf16)fcW[i];
  if (i < (size_t)G4 * HID) {
    int n = (int)(i >> 9), k = (int)(i & 511);
    int g = n >> 9, j = n & 511;
    const float* Ws[4] = {Wf, Wi, Wc, Wo};
    const float* Wg = Ws[g];
    Wh16[i] = (__bf16)Wg[(size_t)j * 1024 + k];
    Wx16[i] = (__bf16)Wg[(size_t)j * 1024 + 512 + k];
  }
  if (i < G4) {
    int g = (int)(i >> 9), j = (int)(i & 511);
    const float* bs[4] = {bf_, bi_, bc_, bo_};
    biascat[i] = bs[g][j];
  }
}

// Embedding lookup -> bf16 x[m=b*T+t][k]
__global__ __launch_bounds__(256) void embed_kernel(
    const int* __restrict__ ids, const float* __restrict__ emb,
    __bf16* __restrict__ x16) {
  size_t i = (size_t)blockIdx.x * blockDim.x + threadIdx.x;  // exact grid
  int m = (int)(i >> 9), k = (int)(i & 511);
  x16[i] = (__bf16)emb[(size_t)ids[m] * HID + k];
}

// ---------------------------------------------------------------------------
// WMMA GEMM with async-LDS A staging: out[m][n] = sum_k A[m][k]*W[n][k]+bias[n]
// K = 512 fixed, M % 64 == 0, N % 256 == 0.
// 256 threads = 8 waves as 2(M) x 4(N); wave tile 32M x 64N; WG tile 64 x 256.
// Per k-step the WG's 64x32 A tile is staged cooperatively into LDS via
// global_load_async_to_lds_b128 (double-buffered, s_wait_asynccnt retired);
// B rows stay direct from global (L2-resident) with global_prefetch ahead.
// ---------------------------------------------------------------------------
__global__ __launch_bounds__(256) void gemm_bf16_kernel(
    const __bf16* __restrict__ A, const __bf16* __restrict__ W,
    const float* __restrict__ bias, float* __restrict__ out, int N) {
  __shared__ __align__(32) __bf16 Abuf[2 * 64 * APAD];  // 2 x (64x32 padded)
  int tid  = threadIdx.x;
  int lane = tid & 31;
  int wave = tid >> 5;
  int m0l  = (wave >> 2) * 32;                // local M offset within WG tile
  int mwg  = blockIdx.x * 64;                 // WG tile M base
  int n0   = blockIdx.y * 256 + (wave & 3) * 64;

  // Cooperative A staging: thread stages 16B of row (tid/4), part (tid%4)
  int srow  = tid >> 2;
  int spart = (tid & 3) * 8;
  const __bf16* sgp = A + (size_t)(mwg + srow) * HID + spart;
  unsigned slds = (unsigned)(uintptr_t)(lds_bf16*)&Abuf[srow * APAD + spart];
  async_copy_b128(slds, sgp);                 // k-step 0 -> buffer 0

  int klo = (lane < 16) ? 0 : 16;
  int kb  = (lane < 16) ? 0 : 16;
  v8f acc[2][4] = {};
  for (int kc = 0; kc < 16; ++kc) {
    int buf = kc & 1;
    if (kc < 15) {
      async_copy_b128(slds + ((kc + 1) & 1) * (64 * APAD * 2), sgp + (kc + 1) * 32);
      asm volatile("s_wait_asynccnt 0x1" ::: "memory");
    } else {
      asm volatile("s_wait_asynccnt 0x0" ::: "memory");
    }
    __syncthreads();  // staged A tile visible to all waves
    int k0 = kc * 32;
    const __bf16* abase = &Abuf[buf * 64 * APAD];
    const __bf16* ap0 = abase + (m0l + (lane & 15)) * APAD + klo;
    v16bf a0 = make_frag(*(const v8bf*)ap0, *(const v8bf*)(ap0 + 8));
    const __bf16* ap1 = ap0 + 16 * APAD;
    v16bf a1 = make_frag(*(const v8bf*)ap1, *(const v8bf*)(ap1 + 8));
#pragma unroll
    for (int ni = 0; ni < 4; ++ni) {
      const __bf16* bp = W + (size_t)(n0 + ni * 16 + (lane & 15)) * HID + k0 + kb;
      v16bf b = *reinterpret_cast<const v16bf*>(bp);
      if (kc < 15) __builtin_prefetch(bp + 32, 0, 3);  // next k-step B row
      acc[0][ni] = WMMA_BF16(a0, b, acc[0][ni]);
      acc[1][ni] = WMMA_BF16(a1, b, acc[1][ni]);
    }
    __syncthreads();  // reads done before this buffer is overwritten
  }

  int hi8 = (lane >= 16) ? 8 : 0;
  int col = lane & 15;
  int m0 = mwg + m0l;
#pragma unroll
  for (int ni = 0; ni < 4; ++ni) {
    int n = n0 + ni * 16 + col;
    float bv = bias[n];
#pragma unroll
    for (int mi = 0; mi < 2; ++mi)
#pragma unroll
      for (int r = 0; r < 8; ++r) {
        int m = m0 + mi * 16 + r + hi8;
        out[(size_t)m * N + n] = acc[mi][ni][r] + bv;
      }
  }
}

// ---------------------------------------------------------------------------
// Persistent LSTM recurrence: single workgroup, 512 threads = 16 waves.
// Wave w owns j-blocks {w*16, w*16+256} x all 4 gates -> gate math in regs.
// h (16x512 bf16) lives in LDS between steps; c lives in registers.
// ---------------------------------------------------------------------------
__global__ __launch_bounds__(512) void lstm_kernel(
    const float* __restrict__ Gx,    // [4096][2048] : bias + x-part preacts
    const __bf16* __restrict__ Wh,   // [2048][512]  : h-part weights (gate concat)
    __bf16* __restrict__ hs16) {     // [4096][512]  : output hidden states
  __shared__ __align__(16) __bf16 hbuf[BATCH * HPAD];
  int tid = threadIdx.x;
  int lane = tid & 31;
  int wave = tid >> 5;
  for (int i = tid; i < BATCH * HPAD; i += 512) hbuf[i] = (__bf16)0.0f;

  int j0[2] = {wave * 16, wave * 16 + 256};
  int hi8 = (lane >= 16) ? 8 : 0;
  int col = lane & 15;
  int klo = (lane < 16) ? 0 : 16;

  v8f acc[2][4] = {};
  float cst[2][8] = {};

  for (int t = 0; t < TLEN; ++t) {
    __syncthreads();  // hbuf (init / previous step) visible
    for (int k0 = 0; k0 < HID; k0 += 32) {
      const __bf16* ap = &hbuf[col * HPAD + k0 + klo];
      v8bf lo = *reinterpret_cast<const v8bf*>(ap);
      v8bf hi = *reinterpret_cast<const v8bf*>(ap + 8);
      v16bf a = make_frag(lo, hi);
#pragma unroll
      for (int jb = 0; jb < 2; ++jb)
#pragma unroll
        for (int g = 0; g < 4; ++g) {
          v16bf b = load_B_frag(Wh, HID, g * 512 + j0[jb], k0, lane);
          acc[jb][g] = WMMA_BF16(a, b, acc[jb][g]);
        }
    }
    __syncthreads();  // all waves done reading hbuf
    v8f vzero = {};
#pragma unroll
    for (int jb = 0; jb < 2; ++jb) {
#pragma unroll
      for (int r = 0; r < 8; ++r) {
        int b = r + hi8;  // batch index
        size_t gbase = ((size_t)(b * TLEN + t)) * G4 + j0[jb] + col;
        float pf = acc[jb][0][r] + Gx[gbase];
        float pi = acc[jb][1][r] + Gx[gbase + 512];
        float pc = acc[jb][2][r] + Gx[gbase + 1024];
        float po = acc[jb][3][r] + Gx[gbase + 1536];
        float fg = 1.0f / (1.0f + __expf(-pf));
        float ig = 1.0f / (1.0f + __expf(-pi));
        float cb = tanhf(pc);
        float c = cst[jb][r] * fg + ig * cb;
        cst[jb][r] = c;
        float h = po * tanhf(c);  // NOTE: no sigmoid on o (faithful to ref)
        __bf16 hb = (__bf16)h;
        hbuf[b * HPAD + j0[jb] + col] = hb;
        hs16[((size_t)(b * TLEN + t)) * HID + j0[jb] + col] = hb;
      }
#pragma unroll
      for (int g = 0; g < 4; ++g) acc[jb][g] = vzero;
    }
  }
}

// ---------------------------------------------------------------------------
// Loss: per-row logsumexp + NLL (deterministic block reductions)
// ---------------------------------------------------------------------------
__global__ __launch_bounds__(256) void loss_rows_kernel(
    const float* __restrict__ logits, const int* __restrict__ tgt,
    float* __restrict__ nll) {
  __shared__ float red[256];
  int row = blockIdx.x, tid = threadIdx.x;
  const float* lg = logits + (size_t)row * VOCAB;
  float mx = -INFINITY;
  for (int v = tid; v < VOCAB; v += 256) mx = fmaxf(mx, lg[v]);
  red[tid] = mx; __syncthreads();
  for (int s = 128; s > 0; s >>= 1) {
    if (tid < s) red[tid] = fmaxf(red[tid], red[tid + s]);
    __syncthreads();
  }
  mx = red[0]; __syncthreads();
  float sum = 0.0f;
  for (int v = tid; v < VOCAB; v += 256) sum += __expf(lg[v] - mx);
  red[tid] = sum; __syncthreads();
  for (int s = 128; s > 0; s >>= 1) {
    if (tid < s) red[tid] += red[tid + s];
    __syncthreads();
  }
  if (tid == 0) {
    float lse = mx + __logf(red[0]);
    nll[row] = lse - lg[tgt[row]];
  }
}

__global__ __launch_bounds__(256) void loss_reduce_kernel(
    const float* __restrict__ nll, float* __restrict__ loss) {
  __shared__ float red[256];
  int tid = threadIdx.x;
  float s = 0.0f;
  for (int i = tid; i < MROWS; i += 256) s += nll[i];
  red[tid] = s; __syncthreads();
  for (int st = 128; st > 0; st >>= 1) {
    if (tid < st) red[tid] += red[tid + st];
    __syncthreads();
  }
  if (tid == 0) *loss = red[0] / (float)MROWS;
}

// ---------------------------------------------------------------------------
extern "C" void kernel_launch(void* const* d_in, const int* in_sizes, int n_in,
                              void* d_out, int out_size, void* d_ws, size_t ws_size,
                              hipStream_t stream) {
  const int*   input_s  = (const int*)d_in[0];
  const int*   output_s = (const int*)d_in[1];
  const float* emb      = (const float*)d_in[2];
  const float* Wf       = (const float*)d_in[3];
  const float* bf_      = (const float*)d_in[4];
  const float* Wi       = (const float*)d_in[5];
  const float* bi_      = (const float*)d_in[6];
  const float* Wc       = (const float*)d_in[7];
  const float* bc_      = (const float*)d_in[8];
  const float* Wo       = (const float*)d_in[9];
  const float* bo_      = (const float*)d_in[10];
  const float* fcW      = (const float*)d_in[11];
  const float* fcb      = (const float*)d_in[12];

  float* logits = (float*)d_out;                       // [4096][32000]
  float* loss   = logits + (size_t)MROWS * VOCAB;      // single float

  char* ws = (char*)d_ws;
  size_t off = 0;
  auto carve = [&](size_t bytes) {
    char* p = ws + off;
    off += (bytes + 255) & ~(size_t)255;
    return p;
  };
  __bf16* Wh16    = (__bf16*)carve((size_t)G4 * HID * 2);
  __bf16* Wx16    = (__bf16*)carve((size_t)G4 * HID * 2);
  float*  biascat = (float*)carve((size_t)G4 * 4);
  __bf16* fcW16   = (__bf16*)carve((size_t)VOCAB * HID * 2);
  __bf16* x16     = (__bf16*)carve((size_t)MROWS * HID * 2);
  float*  Gx      = (float*)carve((size_t)MROWS * G4 * 4);
  __bf16* hs16    = (__bf16*)carve((size_t)MROWS * HID * 2);
  float*  nll     = (float*)carve((size_t)MROWS * 4);

  // 1. bf16 weight prep (fcW dominates grid size: 16.384M elems / 256)
  prep_kernel<<<64000, 256, 0, stream>>>(Wf, Wi, Wc, Wo, bf_, bi_, bc_, bo_,
                                         fcW, Wh16, Wx16, biascat, fcW16);
  // 2. embedding -> bf16 (4096*512 / 256 = 8192)
  embed_kernel<<<8192, 256, 0, stream>>>(input_s, emb, x16);
  // 3. Gx = x @ Wx^T + bias  (M=4096, N=2048)
  gemm_bf16_kernel<<<dim3(MROWS / 64, G4 / 256), 256, 0, stream>>>(
      x16, Wx16, biascat, Gx, G4);
  // 4. sequential LSTM recurrence (single persistent workgroup)
  lstm_kernel<<<1, 512, 0, stream>>>(Gx, Wh16, hs16);
  // 5. logits = hs @ fcW^T + fcb  (M=4096, N=32000) -> d_out
  gemm_bf16_kernel<<<dim3(MROWS / 64, VOCAB / 256), 256, 0, stream>>>(
      hs16, fcW16, fcb, logits, VOCAB);
  // 6. per-row NLL, 7. mean
  loss_rows_kernel<<<MROWS, 256, 0, stream>>>(logits, output_s, nll);
  loss_reduce_kernel<<<1, 256, 0, stream>>>(nll, loss);
}
